// GAELSTMModelWithHourlyHeads_31164282699790
// MI455X (gfx1250) — compile-verified, matrix operand order checked by reading, not run on hardware
//
#include <hip/hip_runtime.h>
#include <hip/hip_bf16.h>

typedef __attribute__((ext_vector_type(16))) _Float16 v16h;
typedef __attribute__((ext_vector_type(8)))  float    v8f;

#define LT 256

// ------------------------- fragment-order packing -----------------------------
// WMMA f16 16x16x32 wave32 operand layout (CDNA5 ISA 7.12.2):
//   lane = half*16 + lm;  per chunk the lane holds 16 f16 with
//   k(i) = kc*32 + half*8 + i + (i>=8 ? 8 : 0),  row/col = lm (+16*tile)
// We store fragments contiguously: frag[tile][kchunk][lane][16], so the GEMM
// does one unconditional 32B load per operand per chunk (global_load_b128 x2,
// contiguous 1KB per wave).

// A: activations f32 [M,K] row-major -> fragment order, zero-padded.
__global__ void k_packA(_Float16* __restrict__ dst, const float* __restrict__ X,
                        int Mm, int Kk, int kchunks) {
  int idx = blockIdx.x * blockDim.x + threadIdx.x;
  int tot = ((Mm + 15) / 16) * kchunks * 512;
  if (idx >= tot) return;
  int i16 = idx & 15;
  int lane = (idx >> 4) & 31;
  int kc = (idx >> 9) % kchunks;
  int mt = idx / (kchunks * 512);
  int half = lane >> 4, lm = lane & 15;
  int row = mt * 16 + lm;
  int k = kc * 32 + half * 8 + i16 + ((i16 >= 8) ? 8 : 0);
  dst[idx] = (row < Mm && k < Kk) ? (_Float16)X[(size_t)row * Kk + k] : (_Float16)0.0f;
}

// B: weights f32 [K,N] row-major -> fragment order, zero-padded.
__global__ void k_packB(_Float16* __restrict__ dst, const float* __restrict__ W,
                        int Kk, int Nn, int kchunks) {
  int idx = blockIdx.x * blockDim.x + threadIdx.x;
  int tot = ((Nn + 15) / 16) * kchunks * 512;
  if (idx >= tot) return;
  int i16 = idx & 15;
  int lane = (idx >> 4) & 31;
  int kc = (idx >> 9) % kchunks;
  int nt = idx / (kchunks * 512);
  int half = lane >> 4, lm = lane & 15;
  int col = nt * 16 + lm;
  int k = kc * 32 + half * 8 + i16 + ((i16 >= 8) ? 8 : 0);
  dst[idx] = (col < Nn && k < Kk) ? (_Float16)W[(size_t)k * Nn + col] : (_Float16)0.0f;
}

// Relation-stacked weight W[r][k][o] (R,K,O) viewed as logical B[K, R*O]
// with B[k, r*O+o] = W[(r*K+k)*O+o]; packed to fragment order.
__global__ void k_packRelB(_Float16* __restrict__ dst, const float* __restrict__ W,
                           int Kk, int Oc, int Rr, int kchunks) {
  int idx = blockIdx.x * blockDim.x + threadIdx.x;
  int Nn = Rr * Oc;
  int tot = ((Nn + 15) / 16) * kchunks * 512;
  if (idx >= tot) return;
  int i16 = idx & 15;
  int lane = (idx >> 4) & 31;
  int kc = (idx >> 9) % kchunks;
  int nt = idx / (kchunks * 512);
  int half = lane >> 4, lm = lane & 15;
  int col = nt * 16 + lm;
  int k = kc * 32 + half * 8 + i16 + ((i16 >= 8) ? 8 : 0);
  _Float16 v = (_Float16)0.0f;
  if (col < Nn && k < Kk) {
    int r = col / Oc, o = col % Oc;
    v = (_Float16)W[((size_t)r * Kk + k) * Oc + o];
  }
  dst[idx] = v;
}

// ------------------------------- WMMA GEMM -----------------------------------
// C[M,N] = A@B from fragment-order operands. 4 waves/block, one 16x16 tile per
// wave, K-loop of (2x b128 A + 2x b128 B + v_wmma).
__global__ void k_gemm(float* __restrict__ C, const _Float16* __restrict__ Af,
                       const _Float16* __restrict__ Bf, int Mm, int Nn,
                       int kchunks, int ntiles) {
  int mt = blockIdx.x;
  int nt = blockIdx.y * 4 + (threadIdx.x >> 5);
  if (nt >= ntiles) return;
  int lane = threadIdx.x & 31;
  int half = lane >> 4, lm = lane & 15;
  const v16h* ap = (const v16h*)(Af + (((size_t)mt * kchunks) * 32 + lane) * 16);
  const v16h* bp = (const v16h*)(Bf + (((size_t)nt * kchunks) * 32 + lane) * 16);
  v8f acc = {};
  for (int kc = 0; kc < kchunks; ++kc) {
    v16h a = ap[(size_t)kc * 32];   // chunk stride = 512 f16 = 32 x 32B
    v16h b = bp[(size_t)kc * 32];
    acc = __builtin_amdgcn_wmma_f32_16x16x32_f16(false, a, false, b, (short)0, acc,
                                                 false, false);
  }
#pragma unroll
  for (int v = 0; v < 8; ++v) {
    int row = mt * 16 + half * 8 + v;
    int col = nt * 16 + lm;
    if (row < Mm && col < Nn) C[(size_t)row * Nn + col] = acc[v];
  }
}

// ------------------------- elementwise kernels --------------------------------

__global__ void k_zero(float* __restrict__ p, int n) {
  int i = blockIdx.x * blockDim.x + threadIdx.x;
  if (i < n) p[i] = 0.0f;
}

__global__ void k_norm(float* __restrict__ out, const float* __restrict__ x,
                       const float* __restrict__ mean, const float* __restrict__ stdv,
                       int nn, int ff) {
  int i = blockIdx.x * blockDim.x + threadIdx.x;
  if (i >= nn * ff) return;
  int f = i % ff;
  out[i] = (x[i] - mean[f]) / (stdv[f] + 1e-8f);
}

__global__ void k_cnt(float* __restrict__ cnt, const int* __restrict__ dst,
                      const float* __restrict__ ea, int Ee, int Rr) {
  int e = blockIdx.x * blockDim.x + threadIdx.x;
  if (e >= Ee) return;
  int r = (int)ea[e * 5 + 4];
  atomicAdd(&cnt[dst[e] * Rr + r], 1.0f);
}

__global__ void k_scatter(float* __restrict__ sums, const float* __restrict__ xw,
                          const int* __restrict__ src, const int* __restrict__ dst,
                          const float* __restrict__ ea, int Ee, int Oc, int Rr) {
  int i = blockIdx.x * blockDim.x + threadIdx.x;
  int per = Oc >> 2;
  int e = i / per;
  if (e >= Ee) return;
  int c = (i % per) << 2;
  int r = (int)ea[e * 5 + 4];
  const float* m = xw + ((size_t)src[e] * Rr + r) * Oc + c;
  float* s = sums + ((size_t)dst[e] * Rr + r) * Oc + c;
  atomicAdd(s + 0, m[0]);
  atomicAdd(s + 1, m[1]);
  atomicAdd(s + 2, m[2]);
  atomicAdd(s + 3, m[3]);
}

// agg(mean over relations) + root + bias -> BN(eval) -> PReLU -> (+residual)
__global__ void k_post(float* __restrict__ out, const float* __restrict__ sums,
                       const float* __restrict__ cnt, const float* __restrict__ rootC,
                       const float* __restrict__ bias, const float* __restrict__ bg,
                       const float* __restrict__ bb, const float* __restrict__ bm,
                       const float* __restrict__ bv, const float* __restrict__ prelu,
                       const float* __restrict__ res, int nn, int Oc, int Rr) {
  int i = blockIdx.x * blockDim.x + threadIdx.x;
  if (i >= nn * Oc) return;
  int n = i / Oc, o = i % Oc;
  float agg = 0.0f;
  for (int r = 0; r < Rr; ++r) {
    float cc = cnt[n * Rr + r];
    agg += sums[((size_t)n * Rr + r) * Oc + o] / fmaxf(cc, 1.0f);
  }
  float v = agg + rootC[i] + bias[o];
  v = (v - bm[o]) * rsqrtf(bv[o] + 1e-5f) * bg[o] + bb[o];
  float a = prelu[0];
  v = v > 0.0f ? v : a * v;
  if (res) v += res[i];
  out[i] = v;
}

__global__ void k_add_bias(float* __restrict__ C, const float* __restrict__ b,
                           int rows, int cols) {
  int i = blockIdx.x * blockDim.x + threadIdx.x;
  if (i < rows * cols) C[i] += b[i % cols];
}

// h = LN(relu(h + b1)) * g + b   (per-row serial; Hd <= 140)
__global__ void k_relu_ln(float* __restrict__ h, const float* __restrict__ bias,
                          const float* __restrict__ g, const float* __restrict__ bb,
                          int rows, int Hd) {
  int row = blockIdx.x * blockDim.x + threadIdx.x;
  if (row >= rows) return;
  float* p = h + (size_t)row * Hd;
  float mean = 0.0f;
  for (int j = 0; j < Hd; ++j) {
    float v = p[j] + bias[j];
    v = v > 0.0f ? v : 0.0f;
    p[j] = v;
    mean += v;
  }
  mean /= (float)Hd;
  float var = 0.0f;
  for (int j = 0; j < Hd; ++j) {
    float d = p[j] - mean;
    var += d * d;
  }
  var /= (float)Hd;
  float rs = rsqrtf(var + 1e-5f);
  for (int j = 0; j < Hd; ++j) p[j] = (p[j] - mean) * rs * g[j] + bb[j];
}

__global__ void k_concat2(float* __restrict__ dst, const float* __restrict__ a,
                          const float* __restrict__ b, int nn, int Ac, int Bc) {
  int i = blockIdx.x * blockDim.x + threadIdx.x;
  int W = Ac + Bc;
  if (i >= nn * W) return;
  int n = i / W, j = i % W;
  dst[i] = (j < Ac) ? a[(size_t)n * Ac + j] : b[(size_t)n * Bc + (j - Ac)];
}

// tiny MLPs (env 64x16, time 12x8): one thread per row, Hd<=16
__global__ void k_small_mlp(float* __restrict__ out, const float* __restrict__ in,
                            const float* __restrict__ W1, const float* __restrict__ b1,
                            const float* __restrict__ g, const float* __restrict__ bb,
                            const float* __restrict__ W2, const float* __restrict__ b2,
                            int rows, int I, int Hd, int Oo) {
  int row = blockIdx.x * blockDim.x + threadIdx.x;
  if (row >= rows) return;
  float hbuf[16];
  const float* x = in + (size_t)row * I;
  float mean = 0.0f;
  for (int j = 0; j < Hd; ++j) {
    float acc = b1[j];
    for (int k = 0; k < I; ++k) acc += x[k] * W1[k * Hd + j];
    acc = acc > 0.0f ? acc : 0.0f;
    hbuf[j] = acc;
    mean += acc;
  }
  mean /= (float)Hd;
  float var = 0.0f;
  for (int j = 0; j < Hd; ++j) {
    float d = hbuf[j] - mean;
    var += d * d;
  }
  var /= (float)Hd;
  float rs = rsqrtf(var + 1e-5f);
  for (int j = 0; j < Hd; ++j) hbuf[j] = (hbuf[j] - mean) * rs * g[j] + bb[j];
  for (int o = 0; o < Oo; ++o) {
    float acc = b2[o];
    for (int j = 0; j < Hd; ++j) acc += hbuf[j] * W2[j * Oo + o];
    out[(size_t)row * Oo + o] = acc;
  }
}

__global__ void k_gather_env(float* __restrict__ envn, const float* __restrict__ emb,
                             const int* __restrict__ bv, int nn) {
  int i = blockIdx.x * blockDim.x + threadIdx.x;
  if (i >= nn * 16) return;
  int n = i / 16, j = i % 16;
  envn[i] = emb[bv[n] * 16 + j];
}

// fused_in[n] = [feat(128) | envn(16) | time_emb_t(8)]
__global__ void k_fused(float* __restrict__ dst, const float* __restrict__ feat,
                        const float* __restrict__ envn, const float* __restrict__ te,
                        int nn) {
  int i = blockIdx.x * blockDim.x + threadIdx.x;
  if (i >= nn * 152) return;
  int n = i / 152, j = i % 152;
  float v;
  if (j < 128) v = feat[(size_t)n * 128 + j];
  else if (j < 144) v = envn[(size_t)n * 16 + (j - 128)];
  else v = te[j - 144];
  dst[i] = v;
}

__global__ void k_lstm(float* __restrict__ h, float* __restrict__ c,
                       const float* __restrict__ g1, const float* __restrict__ g2,
                       const float* __restrict__ b_ih, const float* __restrict__ b_hh,
                       int nn, int Hh) {
  int idx = blockIdx.x * blockDim.x + threadIdx.x;
  if (idx >= nn * Hh) return;
  int n = idx / Hh, j = idx % Hh;
  const float* r1 = g1 + (size_t)n * 4 * Hh;
  const float* r2 = g2 + (size_t)n * 4 * Hh;
  float gi = r1[j] + r2[j] + b_ih[j] + b_hh[j];
  float gf = r1[Hh + j] + r2[Hh + j] + b_ih[Hh + j] + b_hh[Hh + j];
  float gg = r1[2 * Hh + j] + r2[2 * Hh + j] + b_ih[2 * Hh + j] + b_hh[2 * Hh + j];
  float go = r1[3 * Hh + j] + r2[3 * Hh + j] + b_ih[3 * Hh + j] + b_hh[3 * Hh + j];
  float i_ = 1.0f / (1.0f + expf(-gi));
  float f_ = 1.0f / (1.0f + expf(-gf));
  float o_ = 1.0f / (1.0f + expf(-go));
  float cc = f_ * c[idx] + i_ * tanhf(gg);
  c[idx] = cc;
  h[idx] = o_ * tanhf(cc);
}

// pred[n, t] = relu(tmp[n,:]+b1) . W2 + b2
__global__ void k_head(float* __restrict__ out, const float* __restrict__ tmp,
                       const float* __restrict__ b1, const float* __restrict__ W2,
                       const float* __restrict__ b2t, int nn, int t, int Tt) {
  int n = blockIdx.x * blockDim.x + threadIdx.x;
  if (n >= nn) return;
  float acc = 0.0f;
  const float* p = tmp + (size_t)n * 64;
  for (int j = 0; j < 64; ++j) {
    float v = p[j] + b1[j];
    v = v > 0.0f ? v : 0.0f;
    acc += v * W2[j];
  }
  out[(size_t)n * Tt + t] = acc + b2t[0];
}

// ------------------------------- host side -----------------------------------

namespace {
struct BlkP {
  const float *W, *root, *bias, *bn_g, *bn_b, *bn_m, *bn_v, *prelu;
};
struct CgvaeP {
  BlkP enc0, enc1;
  const float *mu_W, *mu_b;
  BlkP dec0, dec1;
  const float *out_W, *out_b;
};
struct MlpP {
  const float *W1, *b1, *ln_g, *ln_b, *W2, *b2;
};
struct CgvaeW {
  _Float16 *e0W, *e0R, *e1W, *e1R, *d0W, *d0R, *d1W, *d1R, *mu, *out;
};

static BlkP parse_blk(void* const* d, int& i) {
  BlkP b;
  b.W = (const float*)d[i++];    b.root = (const float*)d[i++];
  b.bias = (const float*)d[i++]; b.bn_g = (const float*)d[i++];
  b.bn_b = (const float*)d[i++]; b.bn_m = (const float*)d[i++];
  b.bn_v = (const float*)d[i++]; b.prelu = (const float*)d[i++];
  return b;
}
static CgvaeP parse_cgvae(void* const* d, int& i) {
  CgvaeP p;
  p.enc0 = parse_blk(d, i);
  p.enc1 = parse_blk(d, i);
  p.mu_W = (const float*)d[i++]; p.mu_b = (const float*)d[i++];
  p.dec0 = parse_blk(d, i);
  p.dec1 = parse_blk(d, i);
  p.out_W = (const float*)d[i++]; p.out_b = (const float*)d[i++];
  return p;
}
static MlpP parse_mlp(void* const* d, int& i) {
  MlpP p;
  p.W1 = (const float*)d[i++]; p.b1 = (const float*)d[i++];
  p.ln_g = (const float*)d[i++]; p.ln_b = (const float*)d[i++];
  p.W2 = (const float*)d[i++]; p.b2 = (const float*)d[i++];
  return p;
}
static inline int kchunksOf(int K) { return (K + 31) / 32; }
static inline int ntilesOf(int N) { return (N + 15) / 16; }
static inline size_t fragElems(int K, int N) {
  return (size_t)ntilesOf(N) * kchunksOf(K) * 512;
}
}  // namespace

extern "C" void kernel_launch(void* const* d_in, const int* in_sizes, int n_in,
                              void* d_out, int out_size, void* d_ws, size_t ws_size,
                              hipStream_t stream) {
  (void)in_sizes; (void)n_in; (void)out_size; (void)ws_size;

  constexpr int NN = 8000, EE = 128000, TT = 12, FF = 32, HH = 128, LL = 64,
                OO = 128, RR = 5, GG = 64, FUSE_I = 152, FUSE_H = 140, DCAT = LL + FF;
  constexpr int MT = (NN + 15) / 16;  // 500 M-tiles
  constexpr int KC_MAX = 5;           // max K = 152/160 -> 5 chunks

  // ---- parse inputs (setup_inputs dict order, params depth-first dict order) ----
  int ip = 0;
  const float* xs = (const float*)d_in[ip++];          // [13, N, F]
  const int* edge_index = (const int*)d_in[ip++];      // [2, E]
  const float* edge_attr = (const float*)d_in[ip++];   // [E, 5]
  const int* batch_vec = (const int*)d_in[ip++];       // [N]
  const float* env_feats = (const float*)d_in[ip++];   // [G, 16]
  const float* time_feats = (const float*)d_in[ip++];  // [T, 8]
  const float* feat_mean = (const float*)d_in[ip++];
  const float* feat_std = (const float*)d_in[ip++];
  CgvaeP cg0 = parse_cgvae(d_in, ip);
  CgvaeP cgs = parse_cgvae(d_in, ip);
  MlpP h0_enc = parse_mlp(d_in, ip);
  MlpP env_enc = parse_mlp(d_in, ip);
  MlpP time_enc = parse_mlp(d_in, ip);
  MlpP fusion = parse_mlp(d_in, ip);
  const float* Wih = (const float*)d_in[ip++];
  const float* Whh = (const float*)d_in[ip++];
  const float* b_ih = (const float*)d_in[ip++];
  const float* b_hh = (const float*)d_in[ip++];
  const float* head_W1 = (const float*)d_in[ip++];  // [T, 128, 64]
  const float* head_b1 = (const float*)d_in[ip++];  // [T, 64]
  const float* head_W2 = (const float*)d_in[ip++];  // [T, 64]
  const float* head_b2 = (const float*)d_in[ip++];  // [T]

  const int* srcI = edge_index;
  const int* dstI = edge_index + EE;
  float* outF = (float*)d_out;

  // ---- workspace bump allocator ----
  char* base = (char*)d_ws;
  size_t off = 0;
  auto alloc = [&](size_t bytes) -> void* {
    off = (off + 255) & ~(size_t)255;
    void* p = base + off;
    off += bytes;
    return p;
  };
  auto allocF = [&](size_t n) { return (float*)alloc(n * 4); };
  auto allocH = [&](size_t n) { return (_Float16*)alloc(n * 2); };

  auto g1d = [](long long n) { return dim3((unsigned)((n + LT - 1) / LT)); };
  auto zero = [&](float* p, long long n) {
    k_zero<<<g1d(n), LT, 0, stream>>>(p, (int)n);
  };

  // ---- pack all weights to f16 fragment order ----
  auto packB = [&](const float* W, int K, int N) {
    _Float16* p = allocH(fragElems(K, N));
    k_packB<<<g1d((long long)fragElems(K, N)), LT, 0, stream>>>(p, W, K, N, kchunksOf(K));
    return p;
  };
  auto packRelB = [&](const float* W, int K) {
    _Float16* p = allocH(fragElems(K, RR * OO));
    k_packRelB<<<g1d((long long)fragElems(K, RR * OO)), LT, 0, stream>>>(p, W, K, OO, RR,
                                                                         kchunksOf(K));
    return p;
  };

  CgvaeW w0, ws2;
  auto pack_cg = [&](const CgvaeP& p, CgvaeW& w) {
    w.e0W = packRelB(p.enc0.W, FF);
    w.e1W = packRelB(p.enc1.W, HH);
    w.d0W = packRelB(p.dec0.W, DCAT);
    w.d1W = packRelB(p.dec1.W, HH);
    w.e0R = packB(p.enc0.root, FF, OO);
    w.e1R = packB(p.enc1.root, HH, OO);
    w.d0R = packB(p.dec0.root, DCAT, OO);
    w.d1R = packB(p.dec1.root, HH, OO);
    w.mu = packB(p.mu_W, HH, LL);
    w.out = packB(p.out_W, HH, OO);
  };
  pack_cg(cg0, w0);
  pack_cg(cgs, ws2);

  _Float16* h0W1_16 = packB(h0_enc.W1, OO, HH);
  _Float16* h0W2_16 = packB(h0_enc.W2, HH, HH);
  _Float16* fusW1_16 = packB(fusion.W1, FUSE_I, FUSE_H);
  _Float16* fusW2_16 = packB(fusion.W2, FUSE_H, HH);
  _Float16* Wih16 = packB(Wih, HH, 4 * HH);
  _Float16* Whh16 = packB(Whh, HH, 4 * HH);
  const size_t hW1frag = fragElems(HH, 64);
  _Float16* hW1_16 = allocH(hW1frag * TT);
  for (int t = 0; t < TT; ++t)
    k_packB<<<g1d((long long)hW1frag), LT, 0, stream>>>(
        hW1_16 + hW1frag * t, head_W1 + (size_t)t * HH * 64, HH, 64, kchunksOf(HH));

  // ---- activation buffers ----
  float* xnorm = allocF((size_t)NN * FF);
  _Float16* af16 = allocH((size_t)MT * KC_MAX * 512);  // A fragments, max K=160
  float* xw = allocF((size_t)NN * RR * OO);            // also LSTM g1 (N*512 <= N*640)
  float* sums = allocF((size_t)NN * RR * OO);          // also LSTM g2
  float* cnt = allocF((size_t)NN * RR);
  float* rootC = allocF((size_t)NN * OO);
  float* h1 = allocF((size_t)NN * HH);
  float* h2 = allocF((size_t)NN * HH);
  float* mu = allocF((size_t)NN * LL);
  float* dcat = allocF((size_t)NN * DCAT);
  float* d1 = allocF((size_t)NN * HH);
  float* d2 = allocF((size_t)NN * HH);
  float* feat = allocF((size_t)NN * OO);
  float* mtmp = allocF((size_t)NN * FUSE_H);
  float* fusedb = allocF((size_t)NN * FUSE_I);
  float* x_t = allocF((size_t)NN * HH);
  float* h_state = allocF((size_t)NN * HH);
  float* c_state = allocF((size_t)NN * HH);
  float* env_emb = allocF((size_t)GG * 16);
  float* time_emb = allocF((size_t)TT * 8);
  float* envn = allocF((size_t)NN * 16);
  float* headtmp = allocF((size_t)NN * 64);
  float* g1 = xw;   // alias: LSTM gates reuse scatter buffers
  float* g2 = sums;

  // ---- building blocks ----
  auto packA = [&](const float* X, int K) {
    long long tot = (long long)MT * kchunksOf(K) * 512;
    k_packA<<<g1d(tot), LT, 0, stream>>>(af16, X, NN, K, kchunksOf(K));
  };
  auto gemm = [&](float* C, const _Float16* Bf, int Ncol, int K) {
    int nt = ntilesOf(Ncol);
    dim3 grid(MT, (nt + 3) / 4);
    k_gemm<<<grid, dim3(128), 0, stream>>>(C, af16, Bf, NN, Ncol, kchunksOf(K), nt);
  };

  auto run_block = [&](const BlkP& bp, const _Float16* Wp16, const _Float16* R16,
                       const float* xin, int K, float* outb, bool residual) {
    packA(xin, K);
    gemm(xw, Wp16, RR * OO, K);
    zero(sums, (long long)NN * RR * OO);
    zero(cnt, (long long)NN * RR);
    k_cnt<<<g1d(EE), LT, 0, stream>>>(cnt, dstI, edge_attr, EE, RR);
    k_scatter<<<g1d((long long)EE * (OO / 4)), LT, 0, stream>>>(sums, xw, srcI, dstI,
                                                                edge_attr, EE, OO, RR);
    gemm(rootC, R16, OO, K);
    k_post<<<g1d((long long)NN * OO), LT, 0, stream>>>(
        outb, sums, cnt, rootC, bp.bias, bp.bn_g, bp.bn_b, bp.bn_m, bp.bn_v, bp.prelu,
        residual ? xin : (const float*)nullptr, NN, OO, RR);
  };

  auto run_cgvae = [&](const CgvaeP& p, const CgvaeW& w, const float* xn, float* featOut) {
    run_block(p.enc0, w.e0W, w.e0R, xn, FF, h1, false);
    run_block(p.enc1, w.e1W, w.e1R, h1, HH, h2, true);
    packA(h2, HH);
    gemm(mu, w.mu, LL, HH);
    k_add_bias<<<g1d((long long)NN * LL), LT, 0, stream>>>(mu, p.mu_b, NN, LL);
    k_concat2<<<g1d((long long)NN * DCAT), LT, 0, stream>>>(dcat, mu, xn, NN, LL, FF);
    run_block(p.dec0, w.d0W, w.d0R, dcat, DCAT, d1, false);
    run_block(p.dec1, w.d1W, w.d1R, d1, HH, d2, true);
    packA(d2, HH);
    gemm(featOut, w.out, OO, HH);
    k_add_bias<<<g1d((long long)NN * OO), LT, 0, stream>>>(featOut, p.out_b, NN, OO);
  };

  auto run_mlp = [&](const MlpP& p, const _Float16* W1_16, const _Float16* W2_16,
                     const float* in, int I, int Hd, int Oo, float* outb) {
    packA(in, I);
    gemm(mtmp, W1_16, Hd, I);
    k_relu_ln<<<g1d(NN), LT, 0, stream>>>(mtmp, p.b1, p.ln_g, p.ln_b, NN, Hd);
    packA(mtmp, Hd);
    gemm(outb, W2_16, Oo, Hd);
    k_add_bias<<<g1d((long long)NN * Oo), LT, 0, stream>>>(outb, p.b2, NN, Oo);
  };

  // ---- small embeddings ----
  k_small_mlp<<<g1d(GG), LT, 0, stream>>>(env_emb, env_feats, env_enc.W1, env_enc.b1,
                                          env_enc.ln_g, env_enc.ln_b, env_enc.W2,
                                          env_enc.b2, GG, 16, 16, 16);
  k_small_mlp<<<g1d(TT), LT, 0, stream>>>(time_emb, time_feats, time_enc.W1, time_enc.b1,
                                          time_enc.ln_g, time_enc.ln_b, time_enc.W2,
                                          time_enc.b2, TT, 8, 8, 8);
  k_gather_env<<<g1d((long long)NN * 16), LT, 0, stream>>>(envn, env_emb, batch_vec, NN);

  // ---- h0 / c0 ----
  k_norm<<<g1d((long long)NN * FF), LT, 0, stream>>>(xnorm, xs, feat_mean, feat_std, NN, FF);
  run_cgvae(cg0, w0, xnorm, feat);
  run_mlp(h0_enc, h0W1_16, h0W2_16, feat, OO, HH, HH, h_state);
  zero(c_state, (long long)NN * HH);

  // ---- temporal loop: CGVAE -> fusion -> LSTM cell -> head ----
  for (int t = 0; t < TT; ++t) {
    k_norm<<<g1d((long long)NN * FF), LT, 0, stream>>>(
        xnorm, xs + (size_t)(t + 1) * NN * FF, feat_mean, feat_std, NN, FF);
    run_cgvae(cgs, ws2, xnorm, feat);
    k_fused<<<g1d((long long)NN * FUSE_I), LT, 0, stream>>>(fusedb, feat, envn,
                                                            time_emb + t * 8, NN);
    run_mlp(fusion, fusW1_16, fusW2_16, fusedb, FUSE_I, FUSE_H, HH, x_t);

    packA(x_t, HH);
    gemm(g1, Wih16, 4 * HH, HH);
    packA(h_state, HH);
    gemm(g2, Whh16, 4 * HH, HH);
    k_lstm<<<g1d((long long)NN * HH), LT, 0, stream>>>(h_state, c_state, g1, g2, b_ih,
                                                       b_hh, NN, HH);

    packA(h_state, HH);
    gemm(headtmp, hW1_16 + hW1frag * t, 64, HH);
    k_head<<<g1d(NN), LT, 0, stream>>>(outF, headtmp, head_b1 + t * 64, head_W2 + t * 64,
                                       head_b2 + t, NN, t, TT);
  }
}
